// MHA_1185410974340
// MI455X (gfx1250) — matrix operand order repaired and checked by reading, compiled, not run on hardware
//
#include <hip/hip_runtime.h>

// ---------------- problem constants ----------------
constexpr int Bc  = 4;
constexpr int Nn  = 2048;
constexpr int Dd  = 1024;
constexpr int Hh  = 16;
constexpr int HDd = 64;
constexpr int Mrows = Bc * Nn;        // 8192

// GEMM tiling
constexpr int BM = 128, BN = 128, BK = 64;
constexpr int LDK = 72;               // padded LDS row stride (bf16 elems), 144B: 16B aligned

typedef __attribute__((ext_vector_type(8)))  __bf16 v8bf;
typedef __attribute__((ext_vector_type(16))) __bf16 v16bf;
typedef __attribute__((ext_vector_type(8)))  float  v8f;

// ---- CDNA5 async global->LDS path (ASYNCcnt), with portable fallback ----
#if defined(__has_builtin)
#if __has_builtin(__builtin_amdgcn_global_load_async_to_lds_b128) && \
    __has_builtin(__builtin_amdgcn_s_wait_asynccnt)
#define USE_ASYNC_LDS 1
#endif
#endif
#ifndef USE_ASYNC_LDS
#define USE_ASYNC_LDS 0
#endif

#if USE_ASYNC_LDS
typedef int v4i __attribute__((vector_size(16)));
typedef __attribute__((address_space(1))) v4i* g_v4i_p;   // global (prints as __device__)
typedef __attribute__((address_space(3))) v4i* l_v4i_p;   // LDS
#endif

// 16-byte global -> LDS copy (async engine when available)
__device__ __forceinline__ void lds_cp16(void* l, const void* g) {
#if USE_ASYNC_LDS
  __builtin_amdgcn_global_load_async_to_lds_b128(
      (g_v4i_p)(unsigned long long)(size_t)g,
      (l_v4i_p)(unsigned)(size_t)l, 0, 0);
#else
  *(v8bf*)l = *(const v8bf*)g;
#endif
}
__device__ __forceinline__ void async_fence() {
#if USE_ASYNC_LDS
  __builtin_amdgcn_s_wait_asynccnt(0);
#endif
}

__device__ __forceinline__ __bf16 f2bf(float f) {
  unsigned u = __builtin_bit_cast(unsigned, f);
  unsigned r = u + 0x7FFFu + ((u >> 16) & 1u);   // round-to-nearest-even
  unsigned short h = (unsigned short)(r >> 16);
  return __builtin_bit_cast(__bf16, h);
}

// Build a 16-elem bf16 fragment from two 16B-aligned LDS chunks.
__device__ __forceinline__ v16bf frag16(const __bf16* p0, const __bf16* p1) {
  v8bf x0 = *(const v8bf*)p0;
  v8bf x1 = *(const v8bf*)p1;
  v16bf r;
#pragma unroll
  for (int e = 0; e < 8; ++e) { r[e] = x0[e]; r[e + 8] = x1[e]; }
  return r;
}

#define WMMA_BF16(A, Bm, C) \
  __builtin_amdgcn_wmma_f32_16x16x32_bf16(false, (A), false, (Bm), (short)0, (C), false, false)

// ============================================================
// Kernel 1: Y = X @ W^T  (fp32 in, bf16 out, head-split layouts)
// MODE 0: Y[b][h][tok][hd]   (Q, K)
// MODE 1: Y[b][h][hd][tok]   (V transposed)
// ============================================================
template <int MODE>
__global__ __launch_bounds__(256) void k_proj(const float* __restrict__ X,
                                              const float* __restrict__ W,
                                              __bf16* __restrict__ Y) {
  __shared__ __bf16 As[BM * LDK];
  __shared__ __bf16 Bs[BN * LDK];

  const int tid  = threadIdx.x;
  const int lane = tid & 31, wave = tid >> 5;
  const int wm = wave & 3, wn = wave >> 2;          // 4x2 wave grid -> 128x128 block tile
  const int hi = lane >> 4, lo = lane & 15;
  const int m_blk = blockIdx.y * BM, n_blk = blockIdx.x * BN;

  v8f acc[2][4] = {};

  for (int k0 = 0; k0 < Dd; k0 += BK) {
    // stage A (X tile) 128x64 fp32 -> bf16
#pragma unroll
    for (int i = 0; i < 8; ++i) {
      int c = tid + 256 * i;                        // 2048 float4 chunks
      int row = c >> 4, col4 = c & 15;
      float4 f = *(const float4*)(X + (size_t)(m_blk + row) * Dd + k0 + col4 * 4);
      __bf16* d = &As[row * LDK + col4 * 4];
      d[0] = f2bf(f.x); d[1] = f2bf(f.y); d[2] = f2bf(f.z); d[3] = f2bf(f.w);
    }
    // stage B (W rows) 128x64 fp32 -> bf16 ; W is [out][in]: rows are exactly B's [n][k] layout
#pragma unroll
    for (int i = 0; i < 8; ++i) {
      int c = tid + 256 * i;
      int row = c >> 4, col4 = c & 15;
      float4 f = *(const float4*)(W + (size_t)(n_blk + row) * Dd + k0 + col4 * 4);
      __bf16* d = &Bs[row * LDK + col4 * 4];
      d[0] = f2bf(f.x); d[1] = f2bf(f.y); d[2] = f2bf(f.z); d[3] = f2bf(f.w);
    }
    if (k0 + BK < Dd) {
      __builtin_prefetch(X + (size_t)(m_blk + (tid >> 1)) * Dd + k0 + BK, 0, 0);
      __builtin_prefetch(W + (size_t)(n_blk + (tid >> 1)) * Dd + k0 + BK, 0, 0);
    }
    __syncthreads();

#pragma unroll
    for (int kk = 0; kk < BK; kk += 32) {
      v16bf a[2], bm[4];
#pragma unroll
      for (int i = 0; i < 2; ++i) {
        int row = wm * 32 + i * 16 + lo, kb = kk + 8 * hi;
        a[i] = frag16(&As[row * LDK + kb], &As[row * LDK + kb + 16]);
      }
#pragma unroll
      for (int j = 0; j < 4; ++j) {
        int col = wn * 64 + j * 16 + lo, kb = kk + 8 * hi;
        bm[j] = frag16(&Bs[col * LDK + kb], &Bs[col * LDK + kb + 16]);
      }
#pragma unroll
      for (int i = 0; i < 2; ++i)
#pragma unroll
        for (int j = 0; j < 4; ++j) acc[i][j] = WMMA_BF16(a[i], bm[j], acc[i][j]);
    }
    __syncthreads();
  }

#pragma unroll
  for (int i = 0; i < 2; ++i)
#pragma unroll
    for (int j = 0; j < 4; ++j)
#pragma unroll
      for (int r = 0; r < 8; ++r) {
        int gm = m_blk + wm * 32 + i * 16 + r + 8 * hi;   // token row
        int gn = n_blk + wn * 64 + j * 16 + lo;           // feature col
        int b = gm >> 11, tok = gm & (Nn - 1);
        int h = gn >> 6,  hd  = gn & 63;
        float v = acc[i][j][r];
        if (MODE == 0)
          Y[(((size_t)(b * Hh + h) * Nn + tok) << 6) + hd] = f2bf(v);
        else
          Y[((size_t)(b * Hh + h) * HDd + hd) * Nn + tok] = f2bf(v);
      }
}

// ============================================================
// Kernel 2: flash attention, 128-query tiles, 8 waves x 16 queries,
// double-buffered async K/V staging.
// Q/K in [B,H,N,64] bf16 ; V^T in [B,H,64,N] bf16 ; O out [B,N,D] bf16
// ============================================================
__device__ __forceinline__ void stage_kv(const __bf16* Kg, const __bf16* Vg, int kt,
                                         __bf16* Ksb, __bf16* Vsb, int tid) {
#pragma unroll
  for (int i = 0; i < 4; ++i) {               // K tile: 128 x 64, rows = key token
    int c = tid + 256 * i;
    int row = c >> 3, col8 = c & 7;
    lds_cp16(&Ksb[row * 72 + col8 * 8],
             Kg + (size_t)(kt * 128 + row) * 64 + col8 * 8);
  }
#pragma unroll
  for (int i = 0; i < 4; ++i) {               // V^T tile: 64 x 128, rows = hd
    int c = tid + 256 * i;
    int row = c >> 4, col8 = c & 15;
    lds_cp16(&Vsb[row * 136 + col8 * 8],
             Vg + (size_t)row * Nn + kt * 128 + col8 * 8);
  }
}

__global__ __launch_bounds__(256) void k_attn(const __bf16* __restrict__ Qb,
                                              const __bf16* __restrict__ Kb,
                                              const __bf16* __restrict__ Vtb,
                                              const unsigned char* __restrict__ adj,
                                              __bf16* __restrict__ Ob) {
  extern __shared__ __bf16 sm[];
  __bf16* Qs  = sm;                         // [128][72]
  __bf16* Ks0 = Qs  + 128 * 72;             // [128][72] buffer 0
  __bf16* Ks1 = Ks0 + 128 * 72;             // [128][72] buffer 1
  __bf16* Vs0 = Ks1 + 128 * 72;             // [64][136] buffer 0
  __bf16* Vs1 = Vs0 + 64 * 136;             // [64][136] buffer 1
  __bf16* Ps  = Vs1 + 64 * 136;             // [8 waves][16][136]

  const int qt = blockIdx.x, h = blockIdx.y, b = blockIdx.z;
  const int tid = threadIdx.x, lane = tid & 31, wave = tid >> 5;
  const int hi = lane >> 4, lo = lane & 15;
  const size_t bh = (size_t)(b * Hh + h);

  const __bf16* Qg = Qb + bh * Nn * HDd + (size_t)qt * 128 * HDd;
  const __bf16* Kg = Kb + bh * Nn * HDd;
  const __bf16* Vg = Vtb + bh * HDd * Nn;

  // stage Q tile (async) + first K/V tiles (async)
#pragma unroll
  for (int i = 0; i < 4; ++i) {
    int c = tid + 256 * i;
    int row = c >> 3, col8 = c & 7;
    lds_cp16(&Qs[row * 72 + col8 * 8], Qg + (size_t)row * 64 + col8 * 8);
  }
  stage_kv(Kg, Vg, 0, Ks0, Vs0, tid);

  float m_run[8], l_run[8];
  v8f o_acc[4] = {};
#pragma unroll
  for (int r = 0; r < 8; ++r) { m_run[r] = -1e30f; l_run[r] = 0.f; }

  constexpr int NT = Nn / 128;
  for (int kt = 0; kt < NT; ++kt) {
    __bf16* Ksb = (kt & 1) ? Ks1 : Ks0;
    __bf16* Vsb = (kt & 1) ? Vs1 : Vs0;

    async_fence();        // my async writes for this buffer are done
    __syncthreads();      // everyone's writes visible; prior readers finished

    // prefetch next K/V panel into the other buffer while we compute
    if (kt + 1 < NT)
      stage_kv(Kg, Vg, kt + 1, (kt & 1) ? Ks0 : Ks1, (kt & 1) ? Vs0 : Vs1, tid);

    // S = Q K^T : 16 queries x 128 keys per wave
    v8f s[8] = {};
#pragma unroll
    for (int kk = 0; kk < HDd; kk += 32) {
      int arow = wave * 16 + lo, akb = kk + 8 * hi;
      v16bf a = frag16(&Qs[arow * 72 + akb], &Qs[arow * 72 + akb + 16]);
#pragma unroll
      for (int ks = 0; ks < 8; ++ks) {
        int col = ks * 16 + lo, bkb = kk + 8 * hi;
        v16bf bm = frag16(&Ksb[col * 72 + bkb], &Ksb[col * 72 + bkb + 16]);
        s[ks] = WMMA_BF16(a, bm, s[ks]);
      }
    }

    // scale + mask + online softmax
    const int qrow0 = qt * 128 + wave * 16 + 8 * hi;
#pragma unroll
    for (int r = 0; r < 8; ++r) {
      const unsigned char* mp =
          adj + ((size_t)b * Nn + (qrow0 + r)) * Nn + (size_t)kt * 128 + lo;
      float t = -1e30f;
#pragma unroll
      for (int ks = 0; ks < 8; ++ks) {
        float v = s[ks][r] * 0.125f;                 // 1/sqrt(64)
        v = mp[ks * 16] ? v : -1e30f;                // finite sentinel; prefix garbage
        s[ks][r] = v;                                // is annihilated by alpha->0
        t = fmaxf(t, v);
      }
#pragma unroll
      for (int off = 8; off >= 1; off >>= 1) t = fmaxf(t, __shfl_xor(t, off, 32));
      float mnew  = fmaxf(m_run[r], t);
      float alpha = __expf(m_run[r] - mnew);
      m_run[r] = mnew;
      float rs = 0.f;
#pragma unroll
      for (int ks = 0; ks < 8; ++ks) {
        float p = __expf(s[ks][r] - mnew);
        s[ks][r] = p;
        rs += p;
      }
#pragma unroll
      for (int off = 8; off >= 1; off >>= 1) rs += __shfl_xor(rs, off, 32);
      l_run[r] = l_run[r] * alpha + rs;
#pragma unroll
      for (int j = 0; j < 4; ++j) o_acc[j][r] *= alpha;
      // spill P row to wave-private LDS (C-layout -> A-layout via memory)
      __bf16* pr = Ps + (size_t)wave * 16 * 136 + (r + 8 * hi) * 136;
#pragma unroll
      for (int ks = 0; ks < 8; ++ks) pr[ks * 16 + lo] = f2bf(s[ks][r]);
    }

    // O += P @ V  (A = P 16x128, B = V^T tile [hd][key])
    const __bf16* pb = Ps + (size_t)wave * 16 * 136;
#pragma unroll
    for (int k0 = 0; k0 < 128; k0 += 32) {
      int akb = k0 + 8 * hi;
      v16bf a = frag16(&pb[lo * 136 + akb], &pb[lo * 136 + akb + 16]);
#pragma unroll
      for (int j = 0; j < 4; ++j) {
        int col = j * 16 + lo, bkb = k0 + 8 * hi;
        v16bf bm = frag16(&Vsb[col * 136 + bkb], &Vsb[col * 136 + bkb + 16]);
        o_acc[j] = WMMA_BF16(a, bm, o_acc[j]);
      }
    }
  }

  // epilogue: O /= l ; write [B,N,D] bf16
#pragma unroll
  for (int j = 0; j < 4; ++j)
#pragma unroll
    for (int r = 0; r < 8; ++r) {
      int tok  = qt * 128 + wave * 16 + r + 8 * hi;
      int feat = h * 64 + j * 16 + lo;
      float v = o_acc[j][r] / l_run[r];
      Ob[((size_t)b * Nn + tok) * Dd + feat] = f2bf(v);
    }
}

// ============================================================
// Kernel 3: out = O @ Wo^T  (bf16 A via async-LDS, fp32 W -> bf16, fp32 out)
// ============================================================
__global__ __launch_bounds__(256) void k_out(const __bf16* __restrict__ Ob,
                                             const float* __restrict__ W,
                                             float* __restrict__ Y) {
  __shared__ __bf16 As[BM * LDK];
  __shared__ __bf16 Bs[BN * LDK];

  const int tid  = threadIdx.x;
  const int lane = tid & 31, wave = tid >> 5;
  const int wm = wave & 3, wn = wave >> 2;
  const int hi = lane >> 4, lo = lane & 15;
  const int m_blk = blockIdx.y * BM, n_blk = blockIdx.x * BN;

  v8f acc[2][4] = {};

  for (int k0 = 0; k0 < Dd; k0 += BK) {
#pragma unroll
    for (int i = 0; i < 4; ++i) {                   // A: 1024 async 16B chunks
      int c = tid + 256 * i;
      int row = c >> 3, col8 = c & 7;
      lds_cp16(&As[row * LDK + col8 * 8],
               Ob + (size_t)(m_blk + row) * Dd + k0 + col8 * 8);
    }
#pragma unroll
    for (int i = 0; i < 8; ++i) {                   // B: fp32 -> bf16
      int c = tid + 256 * i;
      int row = c >> 4, col4 = c & 15;
      float4 f = *(const float4*)(W + (size_t)(n_blk + row) * Dd + k0 + col4 * 4);
      __bf16* d = &Bs[row * LDK + col4 * 4];
      d[0] = f2bf(f.x); d[1] = f2bf(f.y); d[2] = f2bf(f.z); d[3] = f2bf(f.w);
    }
    async_fence();
    __syncthreads();
#pragma unroll
    for (int kk = 0; kk < BK; kk += 32) {
      v16bf a[2], bm[4];
#pragma unroll
      for (int i = 0; i < 2; ++i) {
        int row = wm * 32 + i * 16 + lo, kb = kk + 8 * hi;
        a[i] = frag16(&As[row * LDK + kb], &As[row * LDK + kb + 16]);
      }
#pragma unroll
      for (int j = 0; j < 4; ++j) {
        int col = wn * 64 + j * 16 + lo, kb = kk + 8 * hi;
        bm[j] = frag16(&Bs[col * LDK + kb], &Bs[col * LDK + kb + 16]);
      }
#pragma unroll
      for (int i = 0; i < 2; ++i)
#pragma unroll
        for (int j = 0; j < 4; ++j) acc[i][j] = WMMA_BF16(a[i], bm[j], acc[i][j]);
    }
    __syncthreads();
  }

#pragma unroll
  for (int i = 0; i < 2; ++i)
#pragma unroll
    for (int j = 0; j < 4; ++j)
#pragma unroll
      for (int r = 0; r < 8; ++r) {
        int gm = m_blk + wm * 32 + i * 16 + r + 8 * hi;
        int gn = n_blk + wn * 64 + j * 16 + lo;
        Y[(size_t)gm * Dd + gn] = acc[i][j][r];
      }
}

// ============================================================
extern "C" void kernel_launch(void* const* d_in, const int* in_sizes, int n_in,
                              void* d_out, int out_size, void* d_ws, size_t ws_size,
                              hipStream_t stream) {
  (void)in_sizes; (void)n_in; (void)out_size; (void)ws_size;
  const float*         x   = (const float*)d_in[0];
  const unsigned char* adj = (const unsigned char*)d_in[1];   // jax bool -> 1 byte
  const float*         Wq  = (const float*)d_in[2];
  const float*         Wk  = (const float*)d_in[3];
  const float*         Wv  = (const float*)d_in[4];
  const float*         Wo  = (const float*)d_in[5];
  float*               out = (float*)d_out;

  // workspace: Q, K, V^T, O  — each B*H*N*64 bf16 = 16.78 MB (total ~67 MB)
  const size_t QSZ = (size_t)Bc * Hh * Nn * HDd;
  __bf16* Qb  = (__bf16*)d_ws;
  __bf16* Kb  = Qb + QSZ;
  __bf16* Vtb = Kb + QSZ;
  __bf16* Obf = Vtb + QSZ;

  dim3 blk(256);
  dim3 gGemm(Dd / BN, Mrows / BM);                  // 8 x 64

  hipLaunchKernelGGL((k_proj<0>), gGemm, blk, 0, stream, x, Wq, Qb);
  hipLaunchKernelGGL((k_proj<0>), gGemm, blk, 0, stream, x, Wk, Kb);
  hipLaunchKernelGGL((k_proj<1>), gGemm, blk, 0, stream, x, Wv, Vtb);

  // dynamic LDS: Q + 2x(K,V^T) buffers + P staging  (~122 KB, ok on 320KB WGP)
  size_t smem = (size_t)(128 * 72 * 3 + 64 * 136 * 2 + 8 * 16 * 136) * sizeof(__bf16);
  hipLaunchKernelGGL(k_attn, dim3(Nn / 128, Hh, Bc), blk, smem, stream,
                     Qb, Kb, Vtb, adj, Obf);

  hipLaunchKernelGGL(k_out, gGemm, blk, 0, stream, Obf, Wo, out);
}
